// CustomTransformerEncoder_9586367004700
// MI455X (gfx1250) — matrix operand order repaired and checked by reading, not compiled, not run on hardware
//
#include <hip/hip_runtime.h>
#include <cmath>

// ---------------------------------------------------------------------------
// Types
// ---------------------------------------------------------------------------
typedef __bf16 bf16_t;
typedef __attribute__((ext_vector_type(16))) __bf16 v16bf;
typedef __attribute__((ext_vector_type(8)))  __bf16 v8bf;
typedef __attribute__((ext_vector_type(8)))  float  v8f;
typedef __attribute__((ext_vector_type(4)))  unsigned int v4u;
typedef int i4v __attribute__((vector_size(16)));   // matches builtin's V4i

// GEMM tiling: block 128x256, 8 waves in 2x4 grid, wave tile 64x64 (4x4 WMMA)
#define BM 128
#define BN 256
#define BK 32
#define LDK 40                 // padded LDS pitch (elems): 80B stride, 16B aligned
#define ABUF (BM * LDK)        // elems per A stage buffer (5120)
#define BBUF (BN * LDK)        // elems per B stage buffer (10240)

// epilogue flags
#define GF_ACC  1
#define GF_RELU 2
#define GF_F32  4
#define GF_BF16 8
#define GF_BIAS 16

// ---------------------------------------------------------------------------
// Helpers
// ---------------------------------------------------------------------------
__device__ __forceinline__ bf16_t f2bf(float f) {
  union { float f; unsigned u; } a; a.f = f;
  unsigned r = a.u + 0x7FFFu + ((a.u >> 16) & 1u);   // round-to-nearest-even
  union { unsigned short s; bf16_t b; } o; o.s = (unsigned short)(r >> 16);
  return o.b;
}

// 16B global -> LDS copy via CDNA5 async-to-LDS (ASYNCcnt-tracked).
__device__ __forceinline__ void cp16(const bf16_t* g, bf16_t* l) {
#if __has_builtin(__builtin_amdgcn_global_load_async_to_lds_b128)
  __builtin_amdgcn_global_load_async_to_lds_b128(
      (__attribute__((address_space(1))) i4v*)(g),
      (__attribute__((address_space(3))) i4v*)(l), 0, 0);
#else
  *(v4u*)l = *(const v4u*)g;
#endif
}

__device__ __forceinline__ void async_wait() {
#if __has_builtin(__builtin_amdgcn_global_load_async_to_lds_b128)
# if __has_builtin(__builtin_amdgcn_s_wait_asynccnt)
  __builtin_amdgcn_s_wait_asynccnt(0);
# else
  asm volatile("s_wait_asynccnt 0" ::: "memory");
# endif
#endif
}

// A fragment: 16x32 bf16. lane<16: row=lane, K 0-7 (v0-3) and 16-23 (v4-7);
// lane>=16: same row group, K 8-15 and 24-31.  (ISA 7.12.2 table)
__device__ __forceinline__ v16bf load_fragA(const bf16_t* base, int lane) {
  const int row = lane & 15;
  const int kh  = (lane >> 4) * 8;
  union { v16bf v; v8bf h[2]; } f;
  f.h[0] = *(const v8bf*)(base + row * LDK + kh);
  f.h[1] = *(const v8bf*)(base + row * LDK + kh + 16);
  return f.v;
}

// B fragment: 32x16 bf16 (we feed row n of W = column n of B).
// lane<16: col=lane, K 0-7 and 16-23; lane>=16: K 8-15 and 24-31.
__device__ __forceinline__ v16bf load_fragB(const bf16_t* base, int lane) {
  const int row = lane & 15;
  const int kh  = (lane >> 4) * 8;
  union { v16bf v; v8bf h[2]; } f;
  f.h[0] = *(const v8bf*)(base + row * LDK + kh);
  f.h[1] = *(const v8bf*)(base + row * LDK + kh + 16);
  return f.v;
}

// ---------------------------------------------------------------------------
// WMMA GEMM:  C[M,N] = epilogue( A[M,K] x B[N,K]^T )
//   A, B bf16 row-major (K contiguous).  Optional: accumulate into existing
//   f32 C, bias (scaled), ReLU, f32 store, bf16 store.
// block = 256 threads (8 waves); block tile 128x256; wave tile 64x64.
// Double-buffered async global->LDS pipeline; 16 WMMA : 16 ds_load per stage.
// ---------------------------------------------------------------------------
__global__ __launch_bounds__(256)
void gemm_bf16_wmma(const bf16_t* __restrict__ A, const bf16_t* __restrict__ B,
                    float* __restrict__ Cf, bf16_t* __restrict__ Cb,
                    const float* __restrict__ bias, float bias_scale,
                    int M, int N, int K, int flags)
{
  __shared__ __align__(16) bf16_t Als[2 * ABUF];   // 20 KB
  __shared__ __align__(16) bf16_t Bls[2 * BBUF];   // 40 KB

  const int tid  = threadIdx.x;
  const int lane = tid & 31;
  const int wid  = tid >> 5;
  const int wm   = (wid & 1) * 64;    // wave M offset in block tile
  const int wn   = (wid >> 1) * 64;   // wave N offset in block tile

  const int bm0 = blockIdx.y * BM;
  const int bn0 = blockIdx.x * BN;

  // global->LDS loader mapping:
  //   A: 256 threads cover 128 rows x 32 cols (2 x 16B chunks / thread)
  //   B: 256 threads cover 256 rows x 32 cols (4 x 16B chunks / thread)
  const int lrowA = tid >> 1;
  const int lcolA = (tid & 1) * 16;
  const bf16_t* Ag = A + (size_t)(bm0 + lrowA) * K + lcolA;
  const bf16_t* Bg = B + (size_t)(bn0 + tid) * K;
  bf16_t* Alb = Als + lrowA * LDK + lcolA;
  bf16_t* Blb = Bls + tid * LDK;

  v8f acc[4][4];
  #pragma unroll
  for (int i = 0; i < 4; ++i)
    #pragma unroll
    for (int j = 0; j < 4; ++j)
      #pragma unroll
      for (int r = 0; r < 8; ++r)
        acc[i][j][r] = 0.f;

  auto load_stage = [&](int kt, int b) {
    const bf16_t* ga = Ag + (size_t)kt * BK;
    const bf16_t* gb = Bg + (size_t)kt * BK;
    bf16_t* la = Alb + b * ABUF;
    bf16_t* lb = Blb + b * BBUF;
    #pragma unroll
    for (int c = 0; c < 2; ++c) cp16(ga + c * 8, la + c * 8);
    #pragma unroll
    for (int c = 0; c < 4; ++c) cp16(gb + c * 8, lb + c * 8);
  };

  const int KT = K / BK;
  load_stage(0, 0);
  async_wait();
  __syncthreads();

  for (int kt = 0; kt < KT; ++kt) {
    const int cur = kt & 1;
    if (kt + 1 < KT) load_stage(kt + 1, cur ^ 1);   // prefetch next stage

    const bf16_t* Ab = Als + cur * ABUF + wm * LDK;
    const bf16_t* Bb = Bls + cur * BBUF + wn * LDK;

    v16bf af[4], bfv[4];
    #pragma unroll
    for (int i = 0; i < 4; ++i) af[i]  = load_fragA(Ab + i * 16 * LDK, lane);
    #pragma unroll
    for (int j = 0; j < 4; ++j) bfv[j] = load_fragB(Bb + j * 16 * LDK, lane);

    #pragma unroll
    for (int i = 0; i < 4; ++i)
      #pragma unroll
      for (int j = 0; j < 4; ++j)
        acc[i][j] = __builtin_amdgcn_wmma_f32_16x16x32_bf16(
            false, af[i], false, bfv[j], (short)0, acc[i][j], false, false);

    async_wait();
    __syncthreads();
  }

  // epilogue.  C/D layout: VGPR r, lane L -> (M = r + 8*(L>=16), N = L&15)
  const bool fACC  = flags & GF_ACC;
  const bool fRELU = flags & GF_RELU;
  const bool fF32  = flags & GF_F32;
  const bool fBF   = flags & GF_BF16;
  const bool fBIAS = flags & GF_BIAS;

  #pragma unroll
  for (int i = 0; i < 4; ++i) {
    const int r0 = bm0 + wm + i * 16 + (lane >> 4) * 8;
    #pragma unroll
    for (int j = 0; j < 4; ++j) {
      const int c = bn0 + wn + j * 16 + (lane & 15);
      const float bv = fBIAS ? bias_scale * bias[c] : 0.f;
      #pragma unroll
      for (int r = 0; r < 8; ++r) {
        const size_t idx = (size_t)(r0 + r) * N + c;
        float v = acc[i][j][r] + bv;
        if (fACC)  v += Cf[idx];
        if (fRELU) v = fmaxf(v, 0.f);
        if (fF32)  Cf[idx] = v;
        if (fBF)   Cb[idx] = f2bf(v);
      }
    }
  }
}

// ---------------------------------------------------------------------------
// LayerNorm over D=1024, one block (256 thr) per row, bf16 output for WMMA.
// ---------------------------------------------------------------------------
__global__ __launch_bounds__(256)
void layernorm_kernel(const float* __restrict__ x, const float* __restrict__ g,
                      const float* __restrict__ b, bf16_t* __restrict__ out)
{
  __shared__ float ps[8], pq[8];
  __shared__ float sM, sR;
  const size_t row = blockIdx.x;
  const int t = threadIdx.x, lane = t & 31, wid = t >> 5;
  const float* xr = x + row * 1024;

  float4 v = ((const float4*)xr)[t];
  float s  = v.x + v.y + v.z + v.w;
  float sq = v.x * v.x + v.y * v.y + v.z * v.z + v.w * v.w;
  #pragma unroll
  for (int o = 16; o; o >>= 1) { s += __shfl_xor(s, o); sq += __shfl_xor(sq, o); }
  if (lane == 0) { ps[wid] = s; pq[wid] = sq; }
  __syncthreads();
  if (t == 0) {
    float S = 0.f, Q = 0.f;
    for (int w = 0; w < 8; ++w) { S += ps[w]; Q += pq[w]; }
    const float mean = S * (1.f / 1024.f);
    const float var  = Q * (1.f / 1024.f) - mean * mean;
    sM = mean; sR = rsqrtf(var + 1e-5f);
  }
  __syncthreads();
  const float m = sM, rs = sR;
  const int c = t * 4;
  bf16_t* orow = out + row * 1024 + c;
  orow[0] = f2bf((v.x - m) * rs * g[c + 0] + b[c + 0]);
  orow[1] = f2bf((v.y - m) * rs * g[c + 1] + b[c + 1]);
  orow[2] = f2bf((v.z - m) * rs * g[c + 2] + b[c + 2]);
  orow[3] = f2bf((v.w - m) * rs * g[c + 3] + b[c + 3]);
}

// ---------------------------------------------------------------------------
// Per-token differential attention (heads attend over head axis, len 32),
// softmax, diff with lambda, out = diff @ v, per-head RMSNorm, scale, bf16.
// One block (128 thr) per token.
// ---------------------------------------------------------------------------
__global__ __launch_bounds__(128)
void diff_attn_kernel(const float* __restrict__ Q, const float* __restrict__ K,
                      const float* __restrict__ V,
                      const float* __restrict__ lq1, const float* __restrict__ lq2,
                      const float* __restrict__ lk1, const float* __restrict__ lk2,
                      bf16_t* __restrict__ Obf, float lambda_init)
{
  __shared__ float qs[32][33], ks[32][33], att[32][33];
  __shared__ float vs[16][64], os[16][64];
  __shared__ float sc[16];
  __shared__ float sLam;

  const int t = threadIdx.x;
  const size_t tok = blockIdx.x;
  const float* Qr = Q + tok * 1024;
  const float* Kr = K + tok * 1024;
  const float* Vr = V + tok * 1024;
  const float scale = 0.17677669529663687f;   // 32^-0.5

  for (int idx = t; idx < 1024; idx += 128) {
    const int j = idx >> 5, d = idx & 31;
    const int h = j & 15, spl = j >> 4;         // q index j = spl*16 + h
    const int ch = h * 64 + d * 2 + spl;        // interleaved channel layout
    qs[j][d] = Qr[ch] * scale;
    ks[j][d] = Kr[ch];
  }
  for (int idx = t; idx < 1024; idx += 128)
    vs[idx >> 6][idx & 63] = Vr[idx];           // v channel = h*64 + d
  if (t == 0) {
    float d1 = 0.f, d2 = 0.f;
    for (int i = 0; i < 32; ++i) { d1 += lq1[i] * lk1[i]; d2 += lq2[i] * lk2[i]; }
    sLam = __expf(d1) - __expf(d2) + lambda_init;
  }
  __syncthreads();

  for (int idx = t; idx < 1024; idx += 128) {
    const int i = idx >> 5, j = idx & 31;
    float s = 0.f;
    #pragma unroll
    for (int d = 0; d < 32; ++d) s += qs[i][d] * ks[j][d];
    att[i][j] = s;
  }
  __syncthreads();

  if (t < 32) {   // softmax row t
    float m = -1e30f;
    for (int j = 0; j < 32; ++j) m = fmaxf(m, att[t][j]);
    float s = 0.f;
    for (int j = 0; j < 32; ++j) { const float e = __expf(att[t][j] - m); att[t][j] = e; s += e; }
    const float inv = 1.f / s;
    for (int j = 0; j < 32; ++j) att[t][j] *= inv;
  }
  __syncthreads();

  const float lam = sLam;
  for (int idx = t; idx < 1024; idx += 128) {
    const int i = idx >> 6, d = idx & 63;       // head i, dim d
    float s = 0.f;
    #pragma unroll
    for (int h = 0; h < 16; ++h)
      s += (att[i][h] - lam * att[16 + i][16 + h]) * vs[h][d];
    os[i][d] = s;
  }
  __syncthreads();

  if (t < 16) {   // per-head RMS scale
    float ss = 0.f;
    for (int d = 0; d < 64; ++d) ss += os[t][d] * os[t][d];
    sc[t] = rsqrtf(ss * (1.f / 64.f) + 1e-5f) * (1.f - lambda_init);
  }
  __syncthreads();

  for (int idx = t; idx < 1024; idx += 128)
    Obf[tok * 1024 + idx] = f2bf(os[idx >> 6][idx & 63] * sc[idx >> 6]);
}

// ---------------------------------------------------------------------------
// Elementwise kernels
// ---------------------------------------------------------------------------
__global__ void cvt_kernel(const float* __restrict__ in, bf16_t* __restrict__ out, size_t n) {
  size_t i = (size_t)blockIdx.x * blockDim.x + threadIdx.x;
  const size_t stride = (size_t)gridDim.x * blockDim.x;
  for (; i < n; i += stride) out[i] = f2bf(in[i]);
}

// rx = bf16( sigmoid(R) * x )
__global__ void rx_kernel(const float* __restrict__ R, const float* __restrict__ x,
                          bf16_t* __restrict__ rx, size_t n) {
  size_t i = (size_t)blockIdx.x * blockDim.x + threadIdx.x;
  const size_t stride = (size_t)gridDim.x * blockDim.x;
  for (; i < n; i += stride) {
    const float r = 1.f / (1.f + __expf(-R[i]));
    rx[i] = f2bf(r * x[i]);
  }
}

// conn = (1-z)*x + z*tanh(G), z = sigmoid(Z); optional bf16 dual store.
__global__ void gate_out_kernel(const float* __restrict__ x, const float* __restrict__ Z,
                                const float* __restrict__ G, float* __restrict__ conn,
                                bf16_t* __restrict__ connb, size_t n) {
  size_t i = (size_t)blockIdx.x * blockDim.x + threadIdx.x;
  const size_t stride = (size_t)gridDim.x * blockDim.x;
  for (; i < n; i += stride) {
    const float z = 1.f / (1.f + __expf(-Z[i]));
    const float o = (1.f - z) * x[i] + z * tanhf(G[i]);
    conn[i] = o;
    if (connb) connb[i] = f2bf(o);
  }
}

// ---------------------------------------------------------------------------
// Host orchestration
// ---------------------------------------------------------------------------
extern "C" void kernel_launch(void* const* d_in, const int* in_sizes, int n_in,
                              void* d_out, int out_size, void* d_ws, size_t ws_size,
                              hipStream_t stream)
{
  (void)in_sizes; (void)n_in; (void)out_size; (void)ws_size;
  const int Tn = 8 * 2048;       // tokens
  const int D  = 1024;
  const int FF = 4096;
  const size_t TD = (size_t)Tn * D;
  const size_t TF = (size_t)Tn * FF;
  const float LAM0 = 0.8f - 0.6f * expf(-0.3f);

  const float* src  = (const float*)d_in[0];
  const float* bo   = (const float*)d_in[5];
  const float* lq1  = (const float*)d_in[6];
  const float* lq2  = (const float*)d_in[7];
  const float* lk1  = (const float*)d_in[8];
  const float* lk2  = (const float*)d_in[9];
  const float* ln1g = (const float*)d_in[10];
  const float* ln1b = (const float*)d_in[11];
  const float* ln2g = (const float*)d_in[12];
  const float* ln2b = (const float*)d_in[13];
  const float* b1   = (const float*)d_in[15];
  const float* b2   = (const float*)d_in[17];
  const float* bg   = (const float*)d_in[24];

  char* wsp = (char*)d_ws;
  size_t off = 0;
  auto take = [&](size_t bytes) -> void* {
    void* p = wsp + off;
    off = (off + bytes + 255) & ~(size_t)255;
    return p;
  };

  bf16_t* wq = (bf16_t*)take((size_t)D * D * 2);
  bf16_t* wk = (bf16_t*)take((size_t)D * D * 2);
  bf16_t* wv = (bf16_t*)take((size_t)D * D * 2);
  bf16_t* wo = (bf16_t*)take((size_t)D * D * 2);
  bf16_t* wr = (bf16_t*)take((size_t)D * D * 2);
  bf16_t* ur = (bf16_t*)take((size_t)D * D * 2);
  bf16_t* wz = (bf16_t*)take((size_t)D * D * 2);
  bf16_t* uz = (bf16_t*)take((size_t)D * D * 2);
  bf16_t* wg = (bf16_t*)take((size_t)D * D * 2);
  bf16_t* ug = (bf16_t*)take((size_t)D * D * 2);
  bf16_t* w1 = (bf16_t*)take((size_t)FF * D * 2);
  bf16_t* w2 = (bf16_t*)take((size_t)FF * D * 2);

  bf16_t* sbf  = (bf16_t*)take(TD * 2);  // ln1 out; later ln2 out
  bf16_t* xbf  = (bf16_t*)take(TD * 2);  // bf16 of gate "x" (src, then conn)
  bf16_t* ybf  = (bf16_t*)take(TD * 2);  // bf16 of gate "y" (relu of src2/src3)
  bf16_t* abf  = (bf16_t*)take(TD * 2);  // attn out bf16; later r*x bf16
  float*  Qf   = (float*)take(TD * 4);   // Q; later gate R pre-act
  float*  Kf   = (float*)take(TD * 4);   // K; later gate Z pre-act
  float*  Vf   = (float*)take(TD * 4);   // V; later gate G pre-act
  float*  conn = (float*)take(TD * 4);
  bf16_t* hbf  = (bf16_t*)take(TF * 2);  // FFN hidden (relu'd) bf16

  auto cvt = [&](const void* in, bf16_t* out, size_t n) {
    cvt_kernel<<<2048, 256, 0, stream>>>((const float*)in, out, n);
  };
  auto gemm = [&](const bf16_t* A, const bf16_t* Bm, float* Cf, bf16_t* Cbp,
                  const float* bias, float bsc, int M, int N, int K, int flags) {
    dim3 g(N / BN, M / BM);
    gemm_bf16_wmma<<<g, dim3(256), 0, stream>>>(A, Bm, Cf, Cbp, bias, bsc, M, N, K, flags);
  };
  // GRU gate: x f32, xb=x bf16, yb=y bf16 -> outF (and optional outB bf16)
  auto gate = [&](const float* x, const bf16_t* xb, const bf16_t* yb,
                  float* outF, bf16_t* outB) {
    gemm(yb, wr, Qf, nullptr, nullptr, 0.f, Tn, D, D, GF_F32);
    gemm(xb, ur, Qf, nullptr, nullptr, 0.f, Tn, D, D, GF_F32 | GF_ACC);
    gemm(yb, wz, Kf, nullptr, nullptr, 0.f, Tn, D, D, GF_F32);
    gemm(xb, uz, Kf, nullptr, bg, -1.f, Tn, D, D, GF_F32 | GF_ACC | GF_BIAS);
    gemm(yb, wg, Vf, nullptr, nullptr, 0.f, Tn, D, D, GF_F32);
    rx_kernel<<<2048, 256, 0, stream>>>(Qf, x, abf, TD);
    gemm(abf, ug, Vf, nullptr, nullptr, 0.f, Tn, D, D, GF_F32 | GF_ACC);
    gate_out_kernel<<<2048, 256, 0, stream>>>(x, Kf, Vf, outF, outB, TD);
  };

  // 0) bf16 conversions (weights + src)
  cvt(d_in[1],  wq, (size_t)D * D);
  cvt(d_in[2],  wk, (size_t)D * D);
  cvt(d_in[3],  wv, (size_t)D * D);
  cvt(d_in[4],  wo, (size_t)D * D);
  cvt(d_in[18], wr, (size_t)D * D);
  cvt(d_in[19], ur, (size_t)D * D);
  cvt(d_in[20], wz, (size_t)D * D);
  cvt(d_in[21], uz, (size_t)D * D);
  cvt(d_in[22], wg, (size_t)D * D);
  cvt(d_in[23], ug, (size_t)D * D);
  cvt(d_in[14], w1, (size_t)FF * D);
  cvt(d_in[16], w2, (size_t)FF * D);
  cvt(src, xbf, TD);

  // 1) ln1 -> s (bf16)
  layernorm_kernel<<<Tn, 256, 0, stream>>>(src, ln1g, ln1b, sbf);

  // 2) Q/K/V projections (f32 out for attention math)
  gemm(sbf, wq, Qf, nullptr, nullptr, 0.f, Tn, D, D, GF_F32);
  gemm(sbf, wk, Kf, nullptr, nullptr, 0.f, Tn, D, D, GF_F32);
  gemm(sbf, wv, Vf, nullptr, nullptr, 0.f, Tn, D, D, GF_F32);

  // 3) per-token differential attention -> abf (bf16)
  diff_attn_kernel<<<Tn, 128, 0, stream>>>(Qf, Kf, Vf, lq1, lq2, lk1, lk2, abf, LAM0);

  // 4) src2 = attn @ Wo^T + bo ; y = relu(src2) stored bf16 only
  gemm(abf, wo, nullptr, ybf, bo, 1.f, Tn, D, D, GF_RELU | GF_BF16 | GF_BIAS);

  // 5) gate 1: conn = GRU(src, y); also conn bf16 into xbf for gate 2
  gate(src, xbf, ybf, conn, xbf);

  // 6) ln2(conn) -> c (bf16, reuse sbf)
  layernorm_kernel<<<Tn, 256, 0, stream>>>(conn, ln2g, ln2b, sbf);

  // 7) FFN: hid = relu(c@W1^T + b1) bf16 ; y2 = relu(hid@W2^T + b2) bf16
  gemm(sbf, w1, nullptr, hbf, b1, 1.f, Tn, FF, D,  GF_RELU | GF_BF16 | GF_BIAS);
  gemm(hbf, w2, nullptr, ybf, b2, 1.f, Tn, D,  FF, GF_RELU | GF_BF16 | GF_BIAS);

  // 8) gate 2: conn2 = GRU(conn, y2) -> d_out (f32)
  gate(conn, xbf, ybf, (float*)d_out, nullptr);
}
